// __FusedMultiheadAttentionTF_83580063580291
// MI455X (gfx1250) — compile-verified, matrix operand order checked by reading, not run on hardware
//
#include <hip/hip_runtime.h>
#include <hip/hip_bf16.h>

// ---------------------------------------------------------------------------
// Fused MHA (token-split variant) for gfx1250 (CDNA5, wave32, WMMA).
// Matmuls: v_wmma_f32_16x16x32_bf16. Flash-softmax attention with
// TDM / async-to-LDS staging of the K tile.
// ---------------------------------------------------------------------------

typedef __bf16 bf16;
typedef __attribute__((ext_vector_type(8)))  __bf16 v8bf;
typedef __attribute__((ext_vector_type(16))) __bf16 v16bf;
typedef __attribute__((ext_vector_type(8)))  float  v8f;
typedef __attribute__((ext_vector_type(4)))  unsigned uint4v;
typedef __attribute__((ext_vector_type(4)))  int     int4v;
typedef __attribute__((ext_vector_type(8)))  int     int8v;

#if __has_builtin(__builtin_amdgcn_tensor_load_to_lds)
#define HAVE_TDM 1
#endif
#if __has_builtin(__builtin_amdgcn_global_load_async_to_lds_b128)
#define HAVE_ASYNC 1
#endif

__device__ __forceinline__ bf16 f2bf(float f) {
    unsigned u = __float_as_uint(f);
    unsigned r = (u + 0x7FFFu + ((u >> 16) & 1u)) >> 16;   // RNE truncation
    unsigned short s = (unsigned short)r;
    bf16 out;
    __builtin_memcpy(&out, &s, 2);
    return out;
}

// 16x32 bf16 WMMA fragment load (A-layout; B uses the mirrored layout from the
// transposed matrix). base -> element (row0, k0), row-major, leading dim ld.
__device__ __forceinline__ v16bf load_frag(const bf16* base, int ld, int lane) {
    const bf16* p = base + (size_t)(lane & 15) * ld + ((lane >> 4) & 1) * 8;
    v8bf lo = *(const v8bf*)(p);
    v8bf hi = *(const v8bf*)(p + 16);
    return __builtin_shufflevector(lo, hi, 0,1,2,3,4,5,6,7,8,9,10,11,12,13,14,15);
}

#define WMMA_BF16(a, b, c) \
    __builtin_amdgcn_wmma_f32_16x16x32_bf16(false, (a), false, (b), (short)0, (c), false, false)

// ---------------------------------------------------------------------------
// Casts
// ---------------------------------------------------------------------------
__global__ void cast_f32_bf16(const float* __restrict__ src, bf16* __restrict__ dst, long long n) {
    long long i = (long long)blockIdx.x * blockDim.x + threadIdx.x;
    long long stride = (long long)gridDim.x * blockDim.x;
    for (; i < n; i += stride) dst[i] = f2bf(src[i]);
}

// src: [K, N] f32 row-major  ->  dst: [N, K] bf16 row-major (i.e. src^T)
__global__ void castT_f32_bf16(const float* __restrict__ src, bf16* __restrict__ dst,
                               int K, int N) {
    long long n = (long long)K * N;
    long long i = (long long)blockIdx.x * blockDim.x + threadIdx.x;
    long long stride = (long long)gridDim.x * blockDim.x;
    for (; i < n; i += stride) {
        int k  = (int)(i / N);
        int nn = (int)(i % N);
        dst[(size_t)nn * K + k] = f2bf(src[i]);
    }
}

// ---------------------------------------------------------------------------
// GEMM: C[M,N] = A[M,K] (bf16, row-major) * BT[N,K]^T (bf16, row-major = B^T)
// Block = 256 thr = 8 waves; block tile 128x128; wave tile 32x64 (2x4 WMMA).
// Fragments stream from global (operands are L2-resident on MI455X's 192MB L2).
// ---------------------------------------------------------------------------
__global__ void gemm_bf16_wmma(const bf16* __restrict__ A, const bf16* __restrict__ BT,
                               float* __restrict__ Cf, bf16* __restrict__ Cb,
                               const float* __restrict__ bias,
                               int K, int lda, int ldbt, int ldc) {
    const int lane = threadIdx.x & 31;
    const int wave = threadIdx.x >> 5;
    const int m0 = blockIdx.y * 128 + (wave >> 1) * 32;
    const int n0 = blockIdx.x * 128 + (wave & 1) * 64;

    v8f acc[2][4];
#pragma unroll
    for (int i = 0; i < 2; ++i)
#pragma unroll
        for (int j = 0; j < 4; ++j) {
            v8f z = {0.f, 0.f, 0.f, 0.f, 0.f, 0.f, 0.f, 0.f};
            acc[i][j] = z;
        }

    const bf16* Arow0 = A + (size_t)m0 * lda;
    const bf16* Arow1 = A + (size_t)(m0 + 16) * lda;

    for (int k = 0; k < K; k += 32) {
        v16bf a0 = load_frag(Arow0 + k, lda, lane);
        v16bf a1 = load_frag(Arow1 + k, lda, lane);
#pragma unroll
        for (int j = 0; j < 4; ++j) {
            v16bf bb = load_frag(BT + (size_t)(n0 + j * 16) * ldbt + k, ldbt, lane);
            acc[0][j] = WMMA_BF16(a0, bb, acc[0][j]);
            acc[1][j] = WMMA_BF16(a1, bb, acc[1][j]);
        }
        if (k + 128 < K) {   // gfx1250 global_prefetch_b8 hints
            __builtin_prefetch(Arow0 + (size_t)(lane & 15) * lda + k + 128, 0, 1);
            __builtin_prefetch(BT + (size_t)(n0 + (lane & 15)) * ldbt + k + 128, 0, 1);
        }
    }

    const int hf = (lane >> 4) & 1;
    const int r  = lane & 15;
#pragma unroll
    for (int i = 0; i < 2; ++i)
#pragma unroll
        for (int j = 0; j < 4; ++j) {
            const int rowb = m0 + i * 16 + hf * 8;
            const int col  = n0 + j * 16 + r;
#pragma unroll
            for (int e = 0; e < 8; ++e) {
                float v = acc[i][j][e];
                if (Cb) Cb[(size_t)(rowb + e) * ldc + col] = f2bf(v);
                else    Cf[(size_t)(rowb + e) * ldc + col] = v + bias[col];
            }
        }
}

// ---------------------------------------------------------------------------
// Flash attention over the token-split heads.
// QKV: bf16 [B*3072, 3072]; row = b*3072 + token, col = h*192 + c.
//   Q: tokens 0..1023, K: 1024..2047, V: 2048..3071 (per batch).
// Out A2: bf16 [B*1024, 3072]; row = b*1024 + q, col = h*192 + c.
// Grid: (qtile=8, h=16, b=8); block 256 = 8 waves; wave owns 16 q-rows.
// K tile staged into LDS by the Tensor Data Mover (fallback: async-to-LDS).
// ---------------------------------------------------------------------------
__global__ void attn_flash_wmma(const bf16* __restrict__ QKV, bf16* __restrict__ A2) {
    __shared__ __align__(16) bf16 Klds[128 * 192];      // K tile, row-major [token][c]
    __shared__ __align__(16) bf16 Vt[192 * 128];        // V tile transposed [c][token]
    __shared__ __align__(16) bf16 Pbuf[8 * 16 * 128];   // per-wave P tiles

    const int lane  = threadIdx.x & 31;
    const int wave  = threadIdx.x >> 5;
    const int tid   = threadIdx.x;
    const int qtile = blockIdx.x;      // 0..7
    const int h     = blockIdx.y;      // 0..15
    const int b     = blockIdx.z;      // 0..7
    const int ch0   = h * 192;
    const int q0    = qtile * 128 + wave * 16;
    const int hf    = (lane >> 4) & 1;
    const int r     = lane & 15;
    const float scale = 0.125f;        // 1/sqrt(64)

    const bf16* Qbase = QKV + ((size_t)(b * 3072 + q0)) * 3072 + ch0;

    float m_i[8], l_i[8];
#pragma unroll
    for (int e = 0; e < 8; ++e) { m_i[e] = -3.0e38f; l_i[e] = 0.f; }
    v8f oacc[12];
#pragma unroll
    for (int nc = 0; nc < 12; ++nc) {
        v8f z = {0.f, 0.f, 0.f, 0.f, 0.f, 0.f, 0.f, 0.f};
        oacc[nc] = z;
    }

    for (int kt = 0; kt < 8; ++kt) {
        __syncthreads();               // all waves done with previous Klds/Vt

        const bf16* Kglob = QKV + ((size_t)(b * 3072 + 1024 + kt * 128)) * 3072 + ch0;
        const bf16* Vglob = QKV + ((size_t)(b * 3072 + 2048 + kt * 128)) * 3072 + ch0;

        // ---- stage K tile (128x192 bf16 = 48KB) into LDS ------------------
#if defined(HAVE_TDM)
        if (wave == 0) {
            unsigned lds_off = (unsigned)(size_t)((__attribute__((address_space(3))) bf16*)Klds);
            unsigned long long ga = (unsigned long long)(size_t)Kglob;
            // D# group 0: count=1 | lds_addr | global_addr[56:0] | type=2
            uint4v g0 = { 1u, lds_off,
                          (unsigned)(ga & 0xFFFFFFFFu),
                          (unsigned)(((ga >> 32) & 0x01FFFFFFu) | (2u << 30)) };
            // D# group 1: data_size=2B; tensor 3072 x 24576; tile 192 x 128;
            //             tensor_dim0_stride = 3072 elements
            int8v g1 = { (int)(1u << 16),        // data_size=1 (2 bytes)
                         (int)(3072u << 16),     // tensor_dim0[15:0]
                         (int)(24576u << 16),    // dim0[31:16]=0 | tensor_dim1[15:0]
                         (int)(192u << 16),      // dim1[31:16]=0 | tile_dim0
                         128,                    // tile_dim1 | tile_dim2=0
                         3072,                   // tensor_dim0_stride[31:0]
                         0, 0 };
            int4v gz4 = { 0, 0, 0, 0 };
#if defined(__clang_major__) && (__clang_major__ >= 23)
            int8v gz8 = { 0, 0, 0, 0, 0, 0, 0, 0 };
            __builtin_amdgcn_tensor_load_to_lds(g0, g1, gz4, gz4, gz8, 0);
#else
            __builtin_amdgcn_tensor_load_to_lds(g0, g1, gz4, gz4, 0);
#endif
        }
#elif defined(HAVE_ASYNC)
        for (int ci = tid; ci < 3072; ci += 256) {   // 3072 x 16B chunks
            int t = ci / 24, cc = (ci % 24) * 8;
            __builtin_amdgcn_global_load_async_to_lds_b128(
                (__attribute__((address_space(1))) void*)(bf16*)(Kglob + (size_t)t * 3072 + cc),
                (__attribute__((address_space(3))) void*)(Klds + t * 192 + cc), 0, 0);
        }
#else
        for (int ci = tid; ci < 3072; ci += 256) {
            int t = ci / 24, cc = (ci % 24) * 8;
            *(v8bf*)(Klds + t * 192 + cc) = *(const v8bf*)(Kglob + (size_t)t * 3072 + cc);
        }
#endif

        // ---- stage V tile transposed: Vt[c][t] (b128 global reads) --------
        for (int idx = tid; idx < 128 * 24; idx += 256) {
            int t = idx / 24, cc = (idx % 24) * 8;
            v8bf tmp = *(const v8bf*)(Vglob + (size_t)t * 3072 + cc);
#pragma unroll
            for (int e = 0; e < 8; ++e) Vt[(cc + e) * 128 + t] = tmp[e];
        }

        // ---- wait for staging, then barrier -------------------------------
#if defined(HAVE_TDM)
        if (wave == 0) {
#if __has_builtin(__builtin_amdgcn_s_wait_tensorcnt)
            __builtin_amdgcn_s_wait_tensorcnt(0);
#else
            asm volatile("s_wait_tensorcnt 0" ::: "memory");
#endif
        }
#elif defined(HAVE_ASYNC)
#if __has_builtin(__builtin_amdgcn_s_wait_asynccnt)
        __builtin_amdgcn_s_wait_asynccnt(0);
#else
        asm volatile("s_wait_asynccnt 0" ::: "memory");
#endif
#endif
        __syncthreads();

        // ---- S = scale * Q K^T : 16 x 128 per wave (K frags from LDS) -----
        v8f s[8];
#pragma unroll
        for (int j = 0; j < 8; ++j) {
            v8f z = {0.f, 0.f, 0.f, 0.f, 0.f, 0.f, 0.f, 0.f};
            s[j] = z;
        }
        for (int kc = 0; kc < 192; kc += 32) {
            v16bf aq = load_frag(Qbase + kc, 3072, lane);
#pragma unroll
            for (int j = 0; j < 8; ++j) {
                v16bf bk = load_frag(Klds + (j * 16) * 192 + kc, 192, lane);
                s[j] = WMMA_BF16(aq, bk, s[j]);
            }
        }

        // ---- online softmax (rows live across 16-lane half-groups) --------
        float alpha[8];
#pragma unroll
        for (int e = 0; e < 8; ++e) {
            float mx = -3.0e38f;
#pragma unroll
            for (int j = 0; j < 8; ++j) { s[j][e] *= scale; mx = fmaxf(mx, s[j][e]); }
#pragma unroll
            for (int off = 1; off < 16; off <<= 1)
                mx = fmaxf(mx, __shfl_xor(mx, off, 32));
            float mn = fmaxf(m_i[e], mx);
            float al = __expf(m_i[e] - mn);
            float rs = 0.f;
#pragma unroll
            for (int j = 0; j < 8; ++j) {
                float p = __expf(s[j][e] - mn);
                s[j][e] = p;
                rs += p;
            }
#pragma unroll
            for (int off = 1; off < 16; off <<= 1)
                rs += __shfl_xor(rs, off, 32);
            l_i[e] = l_i[e] * al + rs;
            m_i[e] = mn;
            alpha[e] = al;
        }
#pragma unroll
        for (int nc = 0; nc < 12; ++nc)
#pragma unroll
            for (int e = 0; e < 8; ++e) oacc[nc][e] *= alpha[e];

        // ---- spill P (bf16) to per-wave LDS region, reload as A-frags -----
#pragma unroll
        for (int j = 0; j < 8; ++j)
#pragma unroll
            for (int e = 0; e < 8; ++e)
                Pbuf[(wave * 16 + hf * 8 + e) * 128 + j * 16 + r] = f2bf(s[j][e]);
        asm volatile("s_wait_dscnt 0" ::: "memory");   // wave-local LDS RAW fence

        // ---- O += P @ V ---------------------------------------------------
#pragma unroll
        for (int kk = 0; kk < 128; kk += 32) {
            v16bf ap = load_frag(Pbuf + wave * 16 * 128 + kk, 128, lane);
#pragma unroll
            for (int nc = 0; nc < 12; ++nc) {
                v16bf bv = load_frag(Vt + (size_t)(nc * 16) * 128 + kk, 128, lane);
                oacc[nc] = WMMA_BF16(ap, bv, oacc[nc]);
            }
        }
    }

    // ---- finalize: O /= l, write bf16 head-interleaved [B*1024, 3072] -----
    float inv[8];
#pragma unroll
    for (int e = 0; e < 8; ++e) inv[e] = 1.0f / l_i[e];
#pragma unroll
    for (int nc = 0; nc < 12; ++nc) {
        const int col = ch0 + nc * 16 + r;
#pragma unroll
        for (int e = 0; e < 8; ++e) {
            size_t orow = (size_t)b * 1024 + qtile * 128 + wave * 16 + hf * 8 + e;
            A2[orow * 3072 + col] = f2bf(oacc[nc][e] * inv[e]);
        }
    }
}

// ---------------------------------------------------------------------------
// Host-side launch
// ---------------------------------------------------------------------------
extern "C" void kernel_launch(void* const* d_in, const int* in_sizes, int n_in,
                              void* d_out, int out_size, void* d_ws, size_t ws_size,
                              hipStream_t stream) {
    (void)in_sizes; (void)n_in; (void)out_size; (void)ws_size;

    const float* x    = (const float*)d_in[0];   // [8,3072,1024]
    const float* wqkv = (const float*)d_in[1];   // [1024,3072]
    const float* dw   = (const float*)d_in[2];   // [3072,1024]
    const float* db   = (const float*)d_in[3];   // [1024]
    float* out = (float*)d_out;                  // [8,1024,1024]

    const long long NX   = 24576LL * 1024;
    const long long NW1  = 1024LL * 3072;
    const long long NW2  = 3072LL * 1024;
    const long long NQKV = 24576LL * 3072;
    const long long NA2  = 8192LL * 3072;

    char* ws = (char*)d_ws;
    bf16* Xbf   = (bf16*)ws;                 ws += NX   * 2;
    bf16* WqkvT = (bf16*)ws;                 ws += NW1  * 2;   // [3072,1024]
    bf16* WdT   = (bf16*)ws;                 ws += NW2  * 2;   // [1024,3072]
    bf16* QKV   = (bf16*)ws;                 ws += NQKV * 2;   // [24576,3072]
    bf16* A2    = (bf16*)ws;                 ws += NA2  * 2;   // [8192,3072]

    cast_f32_bf16<<<2048, 256, 0, stream>>>(x, Xbf, NX);
    castT_f32_bf16<<<1024, 256, 0, stream>>>(wqkv, WqkvT, 1024, 3072);
    castT_f32_bf16<<<1024, 256, 0, stream>>>(dw,   WdT,   3072, 1024);

    // GEMM1: QKV[24576,3072] = Xbf[24576,1024] @ Wqkv   (bf16 out)
    gemm_bf16_wmma<<<dim3(3072 / 128, 24576 / 128), 256, 0, stream>>>(
        Xbf, WqkvT, nullptr, QKV, nullptr, /*K=*/1024, /*lda=*/1024,
        /*ldbt=*/1024, /*ldc=*/3072);

    // Flash attention: A2[8192,3072]
    attn_flash_wmma<<<dim3(8, 16, 8), 256, 0, stream>>>(QKV, A2);

    // GEMM2: out[8192,1024] = A2[8192,3072] @ dense_w + b  (f32 out)
    gemm_bf16_wmma<<<dim3(1024 / 128, 8192 / 128), 256, 0, stream>>>(
        A2, WdT, out, nullptr, db, /*K=*/3072, /*lda=*/3072,
        /*ldbt=*/3072, /*ldc=*/1024);
}
